// Model_26912265077113
// MI455X (gfx1250) — matrix operand order repaired and checked by reading, not compiled, not run on hardware
//
#include <hip/hip_runtime.h>
#include <hip/hip_bf16.h>
#include <stdint.h>

// ---------------------------------------------------------------------------
// Permuted-sequence LSTM (B=128, T=784, H=256) + linear head for gfx1250.
//
// Placement (256 VGPRs/wave cap at 16 waves/WGP, measured via round-1 spills):
//   - W_hh bf16 streamed from WGP$/L2 every step (HBM reads it once);
//     B loads are GLOBAL (LOADcnt only) so they don't couple to LDS waits.
//   - h double-buffered in LDS (2 x 64KB bf16), c in LDS (128KB f32).
//   - Batch processed in 4 runtime-loop quarters; a per-quarter OPAQUE 32-bit
//     offset (asm "+v") stops LICM/CSE from hoisting the 32 mq-invariant B
//     fragments into a 256-VGPR live set (the round-2/3 scratch-spill cause)
//     while preserving the pointer's global address space.
//     Per quarter: acc 4x2 tiles (64 VGPRs) + A-frags + pipelined B ~ 200.
// Per step: 4096 v_wmma_f32_16x16x32_bf16 across 16 waves, 1 barrier.
// ---------------------------------------------------------------------------

typedef __attribute__((ext_vector_type(16))) __bf16 v16bf;
typedef __attribute__((ext_vector_type(8)))  float  v8f;

#define B_  128
#define T_  784
#define H_  256
#define G4  1024   // 4*H

union Frag16 { v16bf v; uint32_t u[8]; };

__device__ __forceinline__ uint16_t f2bf(float f) {
  uint32_t u = __float_as_uint(f);
  uint32_t r = u + 0x7fffu + ((u >> 16) & 1u);   // round-to-nearest-even
  return (uint16_t)(r >> 16);
}
__device__ __forceinline__ float bf2f(uint16_t h) {
  return __uint_as_float(((uint32_t)h) << 16);
}
__device__ __forceinline__ float sigm_(float x)  { return 1.0f / (1.0f + __expf(-x)); }
__device__ __forceinline__ float tanhf_(float x) { return 2.0f / (1.0f + __expf(-2.0f * x)) - 1.0f; }

// ---------------------------------------------------------------------------
// Prep: xT[t][b] = inputs[b][perm[t]]; W16 = bf16(W_hh); biasc = b_ih + b_hh
// ---------------------------------------------------------------------------
__global__ void prep_kernel(const float* __restrict__ x,
                            const int*   __restrict__ perm,
                            const float* __restrict__ W_hh,
                            const float* __restrict__ b_ih,
                            const float* __restrict__ b_hh,
                            uint16_t* __restrict__ W16,
                            float* __restrict__ xT,
                            float* __restrict__ biasc) {
  int i = blockIdx.x * blockDim.x + threadIdx.x;
  if (i < G4 * H_) W16[i] = f2bf(W_hh[i]);
  if (i < T_ * B_) {
    int t = i / B_, b = i % B_;
    xT[i] = x[b * T_ + perm[t]];
  }
  if (i < G4) biasc[i] = b_ih[i] + b_hh[i];
}

// ---------------------------------------------------------------------------
// Persistent LSTM. 512 threads = 16 waves. Wave w owns hidden units
// [16w, 16w+16) across all 4 gates and all 128 batch rows.
// ---------------------------------------------------------------------------
__global__ void __launch_bounds__(512, 1)
lstm_kernel(const uint16_t* __restrict__ W16,
            const float*    __restrict__ xT,
            const float*    __restrict__ biasc,
            const float*    __restrict__ W_ih,
            const float*    __restrict__ lin_W,
            const float*    __restrict__ lin_b,
            float*          __restrict__ out) {
  // 256KB of the WGP's 320KB LDS:
  __shared__ __align__(16) uint16_t hbuf0[B_ * H_];   // 64KB h (even steps read)
  __shared__ __align__(16) uint16_t hbuf1[B_ * H_];   // 64KB h (odd steps read)
  __shared__ __align__(16) float    cls[H_ * B_];     // 128KB c, [unit][batch]

  const int tid   = threadIdx.x;
  const int wave  = tid >> 5;     // 0..15 : unit tile (16 hidden units each)
  const int lane  = tid & 31;
  const int nlane = lane & 15;    // column-within-tile / row-within-tile
  const int khalf = lane >> 4;    // K half-select for A/B fragments

  // ---- zero h0 and c ----
  {
    uint32_t* p = (uint32_t*)hbuf0;
    for (int i = tid; i < (B_ * H_) / 2; i += 512) p[i] = 0u;
    for (int i = tid; i < H_ * B_; i += 512) cls[i] = 0.0f;
  }

  // ---- per-lane gate constants for this unit column ----
  float bias4[4], win4[4];
#pragma unroll
  for (int g = 0; g < 4; ++g) {
    int col = g * H_ + wave * 16 + nlane;
    bias4[g] = biasc[col];
    win4[g]  = W_ih[col];
  }

  const int unit = wave * 16 + nlane;
  // Per-lane B base; gate/kt become immediate offsets on top of it:
  // dword index = g*(H*H/8) + unit*128 + kt*16 + khalf*4
  const uint32_t* pB = (const uint32_t*)W16 + unit * (H_ / 2) + khalf * 4;

  __syncthreads();

  // =========================== time loop ===========================
#pragma unroll 1
  for (int t = 0; t < T_; ++t) {
    const uint16_t* hrd = (t & 1) ? hbuf1 : hbuf0;
    uint16_t*       hwr = (t & 1) ? hbuf0 : hbuf1;

    if (tid == 0 && (t + 8) < T_)
      __builtin_prefetch(xT + (t + 8) * B_, 0, 0);   // global_prefetch_b8

#pragma unroll 1
    for (int mq = 0; mq < 4; ++mq) {                 // batch quarters of 32
      // Opaque zero offset: prevents LICM/CSE of the mq-invariant W16 loads
      // (would form a 256-VGPR live set -> scratch spills) while KEEPING the
      // global address space so B lowers to global_load_b128, not flat.
      uint32_t zoff = 0;
      asm volatile("" : "+v"(zoff));
      const uint32_t* pBq = pB + zoff;

      v8f acc[4][2];
#pragma unroll
      for (int g = 0; g < 4; ++g)
#pragma unroll
        for (int mt = 0; mt < 2; ++mt) acc[g][mt] = (v8f){};

      // A base for this quarter: rows mq*32 .. mq*32+31
      const uint32_t* pA = (const uint32_t*)hrd
                         + (mq * 32 + nlane) * (H_ / 2) + khalf * 4;

#pragma unroll
      for (int kt = 0; kt < 8; ++kt) {               // K tiles of 32
        // A fragments for 2 M-tiles (LDS, bf16).
        // Lane n<16 = row n; dwords 0-3 = K{0..7|8..15}(khalf), 4-7 = +16.
        Frag16 af[2];
#pragma unroll
        for (int mt = 0; mt < 2; ++mt) {
          const uint32_t* p = pA + mt * 16 * (H_ / 2) + kt * 16;
#pragma unroll
          for (int i = 0; i < 4; ++i) af[mt].u[i]     = p[i];
#pragma unroll
          for (int i = 0; i < 4; ++i) af[mt].u[4 + i] = p[8 + i];
        }
#pragma unroll
        for (int g = 0; g < 4; ++g) {
          // Streamed B fragment (WGP$/L2): lane n holds W_hh row (col of B),
          // dwords 0-3 = K{0..7|8..15}(khalf), 4-7 = +16 (matches A).
          Frag16 bf;
          const uint32_t* p = pBq + g * (H_ * H_ / 8) + kt * 16;
#pragma unroll
          for (int i = 0; i < 4; ++i) bf.u[i] = p[i];
#pragma unroll
          for (int i = 0; i < 4; ++i) bf.u[4 + i] = p[8 + i];
#pragma unroll
          for (int mt = 0; mt < 2; ++mt) {
            acc[g][mt] = __builtin_amdgcn_wmma_f32_16x16x32_bf16(
                false, af[mt].v, false, bf.v, (short)0, acc[g][mt], false, false);
          }
        }
      }

      // ---- LSTM cell update: wave-local, c in LDS, write h to back buffer --
#pragma unroll
      for (int mt = 0; mt < 2; ++mt) {
#pragma unroll
        for (int r = 0; r < 8; ++r) {
          int   batch = mq * 32 + mt * 16 + khalf * 8 + r;  // C/D: M=r+8*khalf
          float xv    = xT[t * B_ + batch];
          float gi = acc[0][mt][r] + xv * win4[0] + bias4[0];
          float gf = acc[1][mt][r] + xv * win4[1] + bias4[1];
          float gg = acc[2][mt][r] + xv * win4[2] + bias4[2];
          float go = acc[3][mt][r] + xv * win4[3] + bias4[3];
          float iv = sigm_(gi);
          float fv = sigm_(gf);
          float gv = tanhf_(gg);
          float ov = sigm_(go);
          float cc = fv * cls[unit * B_ + batch] + iv * gv;
          cls[unit * B_ + batch] = cc;
          hwr[batch * H_ + unit] = f2bf(ov * tanhf_(cc));
        }
      }
    }

    __syncthreads();   // new h complete; old buffer free for next step
  }

  // ---- linear head: final h is in hbuf0 (T even) ----
  for (int idx = tid; idx < B_ * 10; idx += 512) {
    int b = idx / 10, c = idx % 10;
    float s = lin_b[c];
    for (int k = 0; k < H_; ++k)
      s += bf2f(hbuf0[b * H_ + k]) * lin_W[c * H_ + k];
    out[idx] = s;
  }
}

// ---------------------------------------------------------------------------
extern "C" void kernel_launch(void* const* d_in, const int* in_sizes, int n_in,
                              void* d_out, int out_size, void* d_ws, size_t ws_size,
                              hipStream_t stream) {
  const float* inputs = (const float*)d_in[0];   // (128, 784)
  const int*   perm   = (const int*)  d_in[1];   // (784,)
  const float* W_ih   = (const float*)d_in[2];   // (1024, 1)
  const float* W_hh   = (const float*)d_in[3];   // (1024, 256)
  const float* b_ih   = (const float*)d_in[4];   // (1024,)
  const float* b_hh   = (const float*)d_in[5];   // (1024,)
  const float* lin_W  = (const float*)d_in[6];   // (10, 256)
  const float* lin_b  = (const float*)d_in[7];   // (10,)

  uint8_t*  ws    = (uint8_t*)d_ws;
  uint16_t* W16   = (uint16_t*)ws;                         // 512 KB
  float*    xT    = (float*)(ws + 524288);                 // 784*128*4 B
  float*    biasc = (float*)(ws + 524288 + 401408);        // 4 KB

  int nprep = G4 * H_;  // 262144 covers all prep ranges
  prep_kernel<<<(nprep + 255) / 256, 256, 0, stream>>>(
      inputs, perm, W_hh, b_ih, b_hh, W16, xT, biasc);

  lstm_kernel<<<1, 512, 0, stream>>>(
      W16, xT, biasc, W_ih, lin_W, lin_b, (float*)d_out);
}
